// DecoderBlock_89017492177538
// MI455X (gfx1250) — matrix operand order repaired and checked by reading, not compiled, not run on hardware
//
#include <hip/hip_runtime.h>
#include <hip/hip_bf16.h>

// ---------------------------------------------------------------------------
// Decoder block for MI455X (gfx1250, wave32, WMMA + async-LDS DMA).
// All matmuls: v_wmma_f32_16x16x32_bf16 (fp32 accumulate). Tile staging:
// global_load_async_to_lds_b128 (ASYNCcnt) with double-buffered LDS so the
// DMA of tile i+1 overlaps the WMMAs of tile i. Weights are converted once
// per call to bf16 *transposed* (NxK) so every stage copy is a contiguous
// row -> pure async DMA, no VGPR pass-through.
// ---------------------------------------------------------------------------

#define EMBED    1024
#define HEADS    16
#define HEAD_DIM 64
#define BATCH    4
#define SEQ      2048
#define FF_DIM   4096
#define MROWS    (BATCH * SEQ)   // 8192

typedef __attribute__((ext_vector_type(16))) __bf16        v16bf;
typedef __attribute__((ext_vector_type(8)))  float         v8f;
typedef __attribute__((ext_vector_type(8)))  unsigned int  v8u;

__device__ __forceinline__ v8f wmma_bf16(v16bf a, v16bf b, v8f c) {
  return __builtin_amdgcn_wmma_f32_16x16x32_bf16(false, a, false, b, (short)0, c,
                                                 false, false);
}

__device__ __forceinline__ unsigned short f32_to_bf16(float f) {
  unsigned int u = __builtin_bit_cast(unsigned int, f);
  unsigned int r = u + 0x7FFFu + ((u >> 16) & 1u);   // RNE
  return (unsigned short)(r >> 16);
}

// ---- CDNA5 async global->LDS DMA (ASYNCcnt-tracked, bypasses VGPRs) -------
__device__ __forceinline__ void async_copy_b128(const unsigned short* gsrc,
                                                unsigned short* ldst) {
  // generic LDS pointer: low 32 bits are the LDS byte offset
  unsigned lds_addr = (unsigned)(size_t)ldst;
  asm volatile("global_load_async_to_lds_b128 %0, %1, off"
               :: "v"(lds_addr), "v"(gsrc)
               : "memory");
}
__device__ __forceinline__ void wait_async0() {
  asm volatile("s_wait_asynccnt 0x0" ::: "memory");
}

// ---------------------------------------------------------------------------
// fp32 -> bf16 TRANSPOSED weight conversion: in K x N (f32) -> out N x K.
// Tiled through LDS so both sides stay coalesced. Block 256 = 32x8.
// ---------------------------------------------------------------------------
__global__ __launch_bounds__(256) void cvt_bf16_t_kernel(
    const float* __restrict__ in, unsigned short* __restrict__ out, int K, int N) {
  __shared__ unsigned short tile[32][33];
  const int bk = blockIdx.x * 32, bn = blockIdx.y * 32;
  const int tx = threadIdx.x & 31, ty = threadIdx.x >> 5;
#pragma unroll
  for (int i = 0; i < 32; i += 8)
    tile[ty + i][tx] = f32_to_bf16(in[(size_t)(bk + ty + i) * N + bn + tx]);
  __syncthreads();
#pragma unroll
  for (int i = 0; i < 32; i += 8)
    out[(size_t)(bn + ty + i) * K + bk + tx] = tile[tx][ty + i];
}

// ---------------------------------------------------------------------------
// LayerNorm over rows of EMBED, bf16 out. One block (256 thr) per row.
// ---------------------------------------------------------------------------
__global__ __launch_bounds__(256) void layernorm_bf16_kernel(
    const float* __restrict__ x, const float* __restrict__ g,
    const float* __restrict__ beta, unsigned short* __restrict__ out) {
  __shared__ float red[256];
  const int row = blockIdx.x;
  const float* xr = x + (size_t)row * EMBED;

  float v[4];
  float s = 0.f;
#pragma unroll
  for (int i = 0; i < 4; ++i) { v[i] = xr[threadIdx.x + i * 256]; s += v[i]; }
  red[threadIdx.x] = s;
  __syncthreads();
  for (int off = 128; off > 0; off >>= 1) {
    if (threadIdx.x < off) red[threadIdx.x] += red[threadIdx.x + off];
    __syncthreads();
  }
  const float mu = red[0] * (1.0f / EMBED);
  __syncthreads();

  float s2 = 0.f;
#pragma unroll
  for (int i = 0; i < 4; ++i) { float d = v[i] - mu; s2 += d * d; }
  red[threadIdx.x] = s2;
  __syncthreads();
  for (int off = 128; off > 0; off >>= 1) {
    if (threadIdx.x < off) red[threadIdx.x] += red[threadIdx.x + off];
    __syncthreads();
  }
  const float rstd = rsqrtf(red[0] * (1.0f / EMBED) + 1e-5f);

#pragma unroll
  for (int i = 0; i < 4; ++i) {
    int c = threadIdx.x + i * 256;
    float y = (v[i] - mu) * rstd * g[c] + beta[c];
    out[(size_t)row * EMBED + c] = f32_to_bf16(y);
  }
}

// ---------------------------------------------------------------------------
// bf16 WMMA GEMM: C = A(MxK) @ Bt(NxK)^T + bias. Epilogues:
//   EPI 0: bf16 store              (Q/K projections)
//   EPI 1: exact-erf GELU -> bf16  (MLP up)
//   EPI 2: fp32 residual add       (attn out proj, MLP down)
//   EPI 3: bf16 store TRANSPOSED per (b,h) as [d][s]  (V projection)
// Block 256 thr (8 waves), tile 128x128, K-step 32, double-buffered async LDS.
// Strides: 40 elems (80 B) -> 16B-aligned rows for async B128, and dword
// stride 20 (gcd(20,64)=4) -> conflict-free 16-lane fragment reads.
// ---------------------------------------------------------------------------
#define GBM 128
#define GBN 128
#define GBK 32
#define ASTR 40
#define BSTR 40

template <int EPI>
__global__ __launch_bounds__(256) void gemm_bf16_kernel(
    const unsigned short* __restrict__ A,    // M x K, bf16
    const unsigned short* __restrict__ Bt,   // N x K, bf16 (transposed)
    const float* __restrict__ bias,          // N
    const float* __restrict__ res,           // M x N (EPI==2)
    unsigned short* __restrict__ Cb,         // bf16 out (EPI 0/1/3)
    float* __restrict__ Cf,                  // f32 out (EPI 2)
    int M, int N, int K) {
  __shared__ __align__(16) unsigned short As[2][GBM * ASTR];
  __shared__ __align__(16) unsigned short Bs[2][GBN * BSTR];

  const int tid  = threadIdx.x;
  const int lane = tid & 31;
  const int wave = tid >> 5;
  const int wm   = wave >> 1;     // 0..3
  const int wn   = wave & 1;      // 0..1
  const int bM   = blockIdx.y * GBM;
  const int bN   = blockIdx.x * GBN;
  const int l15  = lane & 15;
  const int lh   = (lane >= 16) ? 1 : 0;

  auto issue_tile = [&](int bsel, int k0) {
#pragma unroll
    for (int i = 0; i < 2; ++i) {
      int idx = tid + i * 256;          // 0..511
      int row = idx >> 2;               // 0..127
      int col = (idx & 3) * 8;          // 0,8,16,24
      async_copy_b128(A + (size_t)(bM + row) * K + k0 + col,
                      &As[bsel][row * ASTR + col]);
      async_copy_b128(Bt + (size_t)(bN + row) * K + k0 + col,
                      &Bs[bsel][row * BSTR + col]);
    }
  };

  v8f acc[2][4];
#pragma unroll
  for (int i = 0; i < 2; ++i)
#pragma unroll
    for (int j = 0; j < 4; ++j) acc[i][j] = {};

  issue_tile(0, 0);
  int buf = 0;
  for (int k0 = 0; k0 < K; k0 += GBK) {
    wait_async0();          // current tile landed in LDS
    __syncthreads();        // visible to all waves; prev-buf readers done
    if (k0 + GBK < K) issue_tile(buf ^ 1, k0 + GBK);  // overlap next DMA

    v16bf af[2], bfr[4];
#pragma unroll
    for (int mi = 0; mi < 2; ++mi) {
      const unsigned int* p =
          (const unsigned int*)(&As[buf][(wm * 32 + mi * 16 + l15) * ASTR]);
      v8u u;
#pragma unroll
      for (int j = 0; j < 8; ++j) {
        int k = (j < 4 ? 2 * j : 16 + 2 * (j - 4)) + lh * 8;   // A layout
        u[j] = p[k >> 1];
      }
      af[mi] = __builtin_bit_cast(v16bf, u);
    }
#pragma unroll
    for (int ni = 0; ni < 4; ++ni) {
      const unsigned int* p =
          (const unsigned int*)(&Bs[buf][(wn * 64 + ni * 16 + l15) * BSTR]);
      v8u u;
#pragma unroll
      for (int j = 0; j < 8; ++j) {
        int k = (j < 4 ? 2 * j : 16 + 2 * (j - 4)) + lh * 8;   // B layout
        u[j] = p[k >> 1];
      }
      bfr[ni] = __builtin_bit_cast(v16bf, u);
    }
#pragma unroll
    for (int mi = 0; mi < 2; ++mi)
#pragma unroll
      for (int ni = 0; ni < 4; ++ni)
        acc[mi][ni] = wmma_bf16(af[mi], bfr[ni], acc[mi][ni]);
    buf ^= 1;
  }

  // epilogue: C layout row = vgpr r + 8*half, col = lane&15
#pragma unroll
  for (int mi = 0; mi < 2; ++mi) {
#pragma unroll
    for (int ni = 0; ni < 4; ++ni) {
#pragma unroll
      for (int r = 0; r < 8; ++r) {
        int m = bM + wm * 32 + mi * 16 + r + lh * 8;
        int n = bN + wn * 64 + ni * 16 + l15;
        float v = acc[mi][ni][r] + bias[n];
        if (EPI == 0) {
          Cb[(size_t)m * N + n] = f32_to_bf16(v);
        } else if (EPI == 1) {
          float gl = 0.5f * v * (1.0f + erff(v * 0.70710678118654752f));
          Cb[(size_t)m * N + n] = f32_to_bf16(gl);
        } else if (EPI == 2) {
          size_t off = (size_t)m * N + n;
          Cf[off] = res[off] + v;
        } else {  // EPI 3: V^T per (b,h): [d][s]
          size_t off = (((size_t)(m >> 11) * HEADS + (n >> 6)) * HEAD_DIM +
                        (n & 63)) * SEQ + (m & 2047);
          Cb[off] = f32_to_bf16(v);
        }
      }
    }
  }
}

// ---------------------------------------------------------------------------
// Causal flash attention. Block 128 thr (4 waves), 64 query rows per block,
// key blocks of 32 (P 16x32 = one A fragment). K/V tiles double-buffered,
// staged by async DMA (K row-major over d; V from V^T so rows are over s).
// ---------------------------------------------------------------------------
#define KSTR 72   // [key][d] 64+8 pad: 144B rows (16B ok), dword stride 36
#define VSTR 40   // [d][key] 32+8 pad:  80B rows (16B ok), dword stride 20
#define PSTR 34   // per-wave P staging [m][k] (b16 stores, dword reads)

__global__ __launch_bounds__(128) void attention_kernel(
    const unsigned short* __restrict__ Q, const unsigned short* __restrict__ Kb,
    const unsigned short* __restrict__ Vt, unsigned short* __restrict__ Y) {
  __shared__ __align__(16) unsigned short Kls[2][32 * KSTR];
  __shared__ __align__(16) unsigned short Vls[2][HEAD_DIM * VSTR];
  __shared__ unsigned short Pls[4 * 16 * PSTR];

  const int tid  = threadIdx.x;
  const int lane = tid & 31;
  const int wave = tid >> 5;
  const int l15  = lane & 15;
  const int lh   = (lane >= 16) ? 1 : 0;

  const int b     = blockIdx.z;
  const int h     = blockIdx.y;
  const int qblk  = blockIdx.x;
  const int qbase = qblk * 64 + wave * 16;
  const size_t hd = (size_t)h * HEAD_DIM;

  auto issue_kv = [&](int bsel, int jblk) {
    int kb0 = jblk * 32;
#pragma unroll
    for (int i = 0; i < 2; ++i) {
      int idx = tid + i * 128;                       // 0..255
      int key = idx >> 3, col = (idx & 7) * 8;       // 32 x 64
      async_copy_b128(Kb + ((size_t)(b * SEQ) + kb0 + key) * EMBED + hd + col,
                      &Kls[bsel][key * KSTR + col]);
      int dd = idx >> 2, kc = (idx & 3) * 8;         // 64 x 32
      async_copy_b128(Vt + (((size_t)(b * HEADS) + h) * HEAD_DIM + dd) * SEQ +
                          kb0 + kc,
                      &Vls[bsel][dd * VSTR + kc]);
    }
  };

  // Q fragments (16x64 -> two 16x32 A-frags) straight from global
  v16bf qf[2];
  {
    const unsigned int* qp = (const unsigned int*)(
        Q + ((size_t)(b * SEQ) + qbase + l15) * EMBED + hd);
#pragma unroll
    for (int f = 0; f < 2; ++f) {
      v8u u;
#pragma unroll
      for (int j = 0; j < 8; ++j) {
        int k = f * 32 + (j < 4 ? 2 * j : 16 + 2 * (j - 4)) + lh * 8;
        u[j] = qp[k >> 1];
      }
      qf[f] = __builtin_bit_cast(v16bf, u);
    }
  }

  v8f o[4];
#pragma unroll
  for (int i = 0; i < 4; ++i) o[i] = {};
  float rmax[8], rsum[8];
#pragma unroll
  for (int r = 0; r < 8; ++r) { rmax[r] = -1e30f; rsum[r] = 0.f; }

  const int nkb = qblk * 2 + 2;
  issue_kv(0, 0);
  for (int j0 = 0; j0 < nkb; ++j0) {
    const int kbase = j0 * 32;
    const int cur = j0 & 1;
    wait_async0();
    __syncthreads();
    if (j0 + 1 < nkb) issue_kv(cur ^ 1, j0 + 1);   // overlap next K/V DMA

    // --- S = Q K^T : two 16(q)x16(key) tiles, contraction over d=64 ---
    v8f sc[2];
#pragma unroll
    for (int kn = 0; kn < 2; ++kn) {
      v8f a = {};
#pragma unroll
      for (int f = 0; f < 2; ++f) {
        const unsigned int* p =
            (const unsigned int*)(&Kls[cur][(kn * 16 + l15) * KSTR]);
        v8u u;
#pragma unroll
        for (int j = 0; j < 8; ++j) {
          int k = f * 32 + (j < 4 ? 2 * j : 16 + 2 * (j - 4)) + lh * 8;   // B-frag: k=d, n=key
          u[j] = p[k >> 1];
        }
        v16bf kf = __builtin_bit_cast(v16bf, u);
        a = wmma_bf16(qf[f], kf, a);
      }
      sc[kn] = a;
    }

    // --- mask + online softmax ---
    float curmax[8];
#pragma unroll
    for (int r = 0; r < 8; ++r) {
      int qg = qbase + r + lh * 8;
      float m = -1e30f;
#pragma unroll
      for (int kn = 0; kn < 2; ++kn) {
        int kg = kbase + kn * 16 + l15;
        float s = sc[kn][r] * 0.125f;
        s = (kg <= qg) ? s : -1e30f;
        sc[kn][r] = s;
        m = fmaxf(m, s);
      }
#pragma unroll
      for (int sh = 1; sh < 16; sh <<= 1)
        m = fmaxf(m, __shfl_xor(m, sh, 32));
      curmax[r] = m;
    }
#pragma unroll
    for (int r = 0; r < 8; ++r) {
      float mn    = fmaxf(rmax[r], curmax[r]);
      float alpha = __expf(rmax[r] - mn);
      rmax[r]     = mn;
      float ps    = 0.f;
#pragma unroll
      for (int kn = 0; kn < 2; ++kn) {
        float p = __expf(sc[kn][r] - mn);
        sc[kn][r] = p;
        ps += p;
      }
#pragma unroll
      for (int sh = 1; sh < 16; sh <<= 1) ps += __shfl_xor(ps, sh, 32);
      rsum[r] = rsum[r] * alpha + ps;
#pragma unroll
      for (int ni = 0; ni < 4; ++ni) o[ni][r] *= alpha;
    }

    // --- P: C-layout -> bf16 A-layout via per-wave LDS round-trip ---
    unsigned short* pw = Pls + wave * 16 * PSTR;
#pragma unroll
    for (int r = 0; r < 8; ++r) {
      int m = r + lh * 8;
#pragma unroll
      for (int kn = 0; kn < 2; ++kn)
        pw[m * PSTR + kn * 16 + l15] = f32_to_bf16(sc[kn][r]);
    }
    __syncthreads();

    v16bf pf;
    {
      const unsigned int* p = (const unsigned int*)(pw + l15 * PSTR);
      v8u u;
#pragma unroll
      for (int j = 0; j < 8; ++j) {
        int k = (j < 4 ? 2 * j : 16 + 2 * (j - 4)) + lh * 8;
        u[j] = p[k >> 1];
      }
      pf = __builtin_bit_cast(v16bf, u);
    }
    // --- O += P(16x32) @ V(32x64) ---
#pragma unroll
    for (int ni = 0; ni < 4; ++ni) {
      const unsigned int* p =
          (const unsigned int*)(&Vls[cur][(ni * 16 + l15) * VSTR]);
      v8u u;
#pragma unroll
      for (int j = 0; j < 8; ++j) {
        int k = (j < 4 ? 2 * j : 16 + 2 * (j - 4)) + lh * 8;   // B-frag: k=key, n=d
        u[j] = p[k >> 1];
      }
      v16bf vf = __builtin_bit_cast(v16bf, u);
      o[ni] = wmma_bf16(pf, vf, o[ni]);
    }
  }

  // --- normalize and store bf16 ---
#pragma unroll
  for (int r = 0; r < 8; ++r) {
    float inv = 1.0f / rsum[r];
    int q = qbase + r + lh * 8;
#pragma unroll
    for (int ni = 0; ni < 4; ++ni) {
      int dd = ni * 16 + l15;
      Y[((size_t)(b * SEQ) + q) * EMBED + hd + dd] = f32_to_bf16(o[ni][r] * inv);
    }
  }
}

// ---------------------------------------------------------------------------
// Workspace layout (bytes)
// ---------------------------------------------------------------------------
static const size_t SZ_WE  = (size_t)EMBED * EMBED * 2;
static const size_t SZ_WF  = (size_t)EMBED * FF_DIM * 2;
static const size_t SZ_ACT = (size_t)MROWS * EMBED * 2;
static const size_t OFF_WQ = 0;
static const size_t OFF_WK = OFF_WQ + SZ_WE;
static const size_t OFF_WV = OFF_WK + SZ_WE;
static const size_t OFF_WO = OFF_WV + SZ_WE;
static const size_t OFF_W1 = OFF_WO + SZ_WE;
static const size_t OFF_W2 = OFF_W1 + SZ_WF;
static const size_t OFF_H  = OFF_W2 + SZ_WF;
static const size_t OFF_Q  = OFF_H + SZ_ACT;
static const size_t OFF_K  = OFF_Q + SZ_ACT;
static const size_t OFF_V  = OFF_K + SZ_ACT;   // holds V^T [b][h][d][s]
static const size_t OFF_Y  = OFF_V + SZ_ACT;
static const size_t OFF_X1 = OFF_Y + SZ_ACT;
static const size_t OFF_FF = OFF_X1 + (size_t)MROWS * EMBED * 4;

extern "C" void kernel_launch(void* const* d_in, const int* in_sizes, int n_in,
                              void* d_out, int out_size, void* d_ws,
                              size_t ws_size, hipStream_t stream) {
  (void)in_sizes; (void)n_in; (void)out_size; (void)ws_size;
  const float* x     = (const float*)d_in[0];
  const float* Wq    = (const float*)d_in[1];
  const float* bq    = (const float*)d_in[2];
  const float* Wk    = (const float*)d_in[3];
  const float* bk    = (const float*)d_in[4];
  const float* Wv    = (const float*)d_in[5];
  const float* bv    = (const float*)d_in[6];
  const float* Wo    = (const float*)d_in[7];
  const float* bo    = (const float*)d_in[8];
  const float* g1    = (const float*)d_in[9];
  const float* beta1 = (const float*)d_in[10];
  const float* g2    = (const float*)d_in[11];
  const float* beta2 = (const float*)d_in[12];
  const float* W1    = (const float*)d_in[13];
  const float* b1    = (const float*)d_in[14];
  const float* W2    = (const float*)d_in[15];
  const float* b2    = (const float*)d_in[16];
  float* out = (float*)d_out;

  char* ws = (char*)d_ws;
  unsigned short* wq = (unsigned short*)(ws + OFF_WQ);
  unsigned short* wk = (unsigned short*)(ws + OFF_WK);
  unsigned short* wv = (unsigned short*)(ws + OFF_WV);
  unsigned short* wo = (unsigned short*)(ws + OFF_WO);
  unsigned short* w1 = (unsigned short*)(ws + OFF_W1);
  unsigned short* w2 = (unsigned short*)(ws + OFF_W2);
  unsigned short* hb = (unsigned short*)(ws + OFF_H);
  unsigned short* qb = (unsigned short*)(ws + OFF_Q);
  unsigned short* kb = (unsigned short*)(ws + OFF_K);
  unsigned short* vt = (unsigned short*)(ws + OFF_V);
  unsigned short* yb = (unsigned short*)(ws + OFF_Y);
  float*          x1 = (float*)(ws + OFF_X1);
  unsigned short* fb = (unsigned short*)(ws + OFF_FF);

  // 1) weights -> bf16, transposed to N x K (coalesced tiled transpose)
  {
    dim3 gEEt(EMBED / 32, EMBED / 32);
    cvt_bf16_t_kernel<<<gEEt, 256, 0, stream>>>(Wq, wq, EMBED, EMBED);
    cvt_bf16_t_kernel<<<gEEt, 256, 0, stream>>>(Wk, wk, EMBED, EMBED);
    cvt_bf16_t_kernel<<<gEEt, 256, 0, stream>>>(Wv, wv, EMBED, EMBED);
    cvt_bf16_t_kernel<<<gEEt, 256, 0, stream>>>(Wo, wo, EMBED, EMBED);
    dim3 g1t(EMBED / 32, FF_DIM / 32);
    cvt_bf16_t_kernel<<<g1t, 256, 0, stream>>>(W1, w1, EMBED, FF_DIM);
    dim3 g2t(FF_DIM / 32, EMBED / 32);
    cvt_bf16_t_kernel<<<g2t, 256, 0, stream>>>(W2, w2, FF_DIM, EMBED);
  }

  // 2) LN1
  layernorm_bf16_kernel<<<MROWS, 256, 0, stream>>>(x, g1, beta1, hb);

  // 3) Q/K/V projections (V stored transposed per-(b,h) for async V tiles)
  dim3 gEE(EMBED / GBN, MROWS / GBM);
  gemm_bf16_kernel<0><<<gEE, 256, 0, stream>>>(hb, wq, bq, nullptr, qb, nullptr,
                                               MROWS, EMBED, EMBED);
  gemm_bf16_kernel<0><<<gEE, 256, 0, stream>>>(hb, wk, bk, nullptr, kb, nullptr,
                                               MROWS, EMBED, EMBED);
  gemm_bf16_kernel<3><<<gEE, 256, 0, stream>>>(hb, wv, bv, nullptr, vt, nullptr,
                                               MROWS, EMBED, EMBED);

  // 4) causal flash attention
  dim3 ga(SEQ / 64, HEADS, BATCH);
  attention_kernel<<<ga, 128, 0, stream>>>(qb, kb, vt, yb);

  // 5) x1 = x + y@Wo + bo
  gemm_bf16_kernel<2><<<gEE, 256, 0, stream>>>(yb, wo, bo, x, nullptr, x1,
                                               MROWS, EMBED, EMBED);

  // 6) LN2
  layernorm_bf16_kernel<<<MROWS, 256, 0, stream>>>(x1, g2, beta2, hb);

  // 7) MLP
  dim3 gEF(FF_DIM / GBN, MROWS / GBM);
  gemm_bf16_kernel<1><<<gEF, 256, 0, stream>>>(hb, w1, b1, nullptr, fb, nullptr,
                                               MROWS, FF_DIM, EMBED);
  gemm_bf16_kernel<2><<<gEE, 256, 0, stream>>>(fb, w2, b2, x1, nullptr, out,
                                               MROWS, EMBED, FF_DIM);
}
